// transformer_layer_54219667145429
// MI455X (gfx1250) — compile-verified, matrix operand order checked by reading, not run on hardware
//
#include <hip/hip_runtime.h>
#include <math.h>

typedef __attribute__((ext_vector_type(16))) _Float16 v16h;
typedef __attribute__((ext_vector_type(8)))  _Float16 v8h;
typedef __attribute__((ext_vector_type(8)))  float    v8f;

#define NB     2
#define NPTS   512
#define CH     64
#define EPSV   1e-5f
#define JSPLIT 4

// ---- workspace layout (float offsets) ----
#define OFF_Q    0          // [NB*NPTS*64]
#define OFF_K    65536
#define OFF_V    131072
#define OFF_QB   196608
#define OFF_KB   262144
#define OFF_PART 327680     // NB*NPTS*JSPLIT*192 = 786432
#define SB       1114112
#define OFF_S1   (SB+0)     // 64  sum delta
#define OFF_S2   (SB+64)    // 64  sum delta^2
#define OFF_SQD  (SB+128)   // 64  sum q*delta
#define OFF_SKD  (SB+192)   // 64  sum k*delta
#define OFF_AQ   (SB+256)   // [2][64] per-batch sum q
#define OFF_AK   (SB+384)
#define OFF_AQ2  (SB+512)
#define OFF_AK2  (SB+640)
#define OFF_SP   (SB+768)   // [2][3]  pos sums
#define OFF_SPP  (SB+774)   // [2][9]  pos outer-product sums
#define OFF_A1   (SB+800)   // 9  folded BN1*Wp1
#define OFF_C1   (SB+809)   // 3  folded BN1 shift (= b_p)
#define OFF_S2C  (SB+816)   // 64 bn2 scale
#define OFF_T2C  (SB+880)   // 64 bn2 shift
#define OFF_W2P  (SB+944)   // 64*4 packed (Wp2, bp2)
#define OFF_W2S  (SB+1200)  // 64*4 packed scaled (s2*Wp2, s2*bp2)
#define STATS_ZERO_COUNT 792

__device__ inline v8f wmma_f16(v16h a, v16h b, v8f c) {
    return __builtin_amdgcn_wmma_f32_16x16x32_f16(false, a, false, b, (short)0, c, false, false);
}

// Fetch one WMMA A/B operand (f16, K=32 slice) from a row-major [rows][64] LDS tile.
// lane<16 holds K {base..base+7, base+16..base+23}; lane>=16 holds {+8} of each run.
__device__ inline v16h lds_load_ab(const _Float16* rowptr, int lane, int ks) {
    int off = (lane < 16) ? 0 : 8;
    v8h a0 = *(const v8h*)(rowptr + ks * 32 + off);
    v8h a1 = *(const v8h*)(rowptr + ks * 32 + 16 + off);
    v16h r;
#pragma unroll
    for (int t = 0; t < 8; ++t) { r[t] = a0[t]; r[t + 8] = a1[t]; }
    return r;
}

// load 4 runs of 8 contiguous floats (WMMA-A channel pattern) from a 64-float row
__device__ inline void load_runs(const float* rowp, int off, float* arr) {
#pragma unroll
    for (int ks = 0; ks < 2; ++ks)
#pragma unroll
        for (int g = 0; g < 2; ++g) {
            int bcol = ks * 32 + g * 16 + off;
            float4 u0 = *(const float4*)(rowp + bcol);
            float4 u1 = *(const float4*)(rowp + bcol + 4);
            int d = ks * 16 + g * 8;
            arr[d + 0] = u0.x; arr[d + 1] = u0.y; arr[d + 2] = u0.z; arr[d + 3] = u0.w;
            arr[d + 4] = u1.x; arr[d + 5] = u1.y; arr[d + 6] = u1.z; arr[d + 7] = u1.w;
        }
}

__global__ void k_zero(float* ws) {
    int i = blockIdx.x * blockDim.x + threadIdx.x;
    if (i < STATS_ZERO_COUNT) ws[SB + i] = 0.f;
}

// q,k,v = cf @ W^T + b via WMMA; also per-batch sums/sumsq of q,k.
// ks-outer / 4 live accumulators => consecutive WMMAs independent (no hazard NOPs).
__global__ __launch_bounds__(32) void k_qkv(const float* cf,
        const float* Wq, const float* bq, const float* Wk, const float* bk,
        const float* Wv, const float* bv, float* ws) {
    __shared__ __align__(16) _Float16 wL[3][4096];
    __shared__ __align__(16) _Float16 cfT[1024];
    int lane = threadIdx.x;
    const float* Wsrc[3] = { Wq, Wk, Wv };
    for (int o = 0; o < 3; ++o)
        for (int t = 0; t < 128; ++t) { int idx = t * 32 + lane; wL[o][idx] = (_Float16)Wsrc[o][idx]; }
    int r0 = blockIdx.x * 16;
    for (int t = 0; t < 32; ++t) { int idx = t * 32 + lane; cfT[idx] = (_Float16)cf[r0 * 64 + idx]; }
    __syncthreads();
    int b = r0 / NPTS;
    int col = lane & 15;
    int rbase = 8 * (lane >> 4);
    v16h a0 = lds_load_ab(cfT + col * 64, lane, 0);
    v16h a1 = lds_load_ab(cfT + col * 64, lane, 1);
    const float* bias[3] = { bq, bk, bv };
    float* outp[3]  = { ws + OFF_Q,  ws + OFF_K,  ws + OFF_V };
    float* sum1p[2] = { ws + OFF_AQ, ws + OFF_AK };
    float* sum2p[2] = { ws + OFF_AQ2, ws + OFF_AK2 };
    for (int o = 0; o < 3; ++o) {
        v8f acc[4];
#pragma unroll
        for (int nt = 0; nt < 4; ++nt) {
            float bv0 = bias[o][nt * 16 + col];
#pragma unroll
            for (int t = 0; t < 8; ++t) acc[nt][t] = bv0;
        }
#pragma unroll
        for (int nt = 0; nt < 4; ++nt) {
            v16h bm = lds_load_ab(wL[o] + (nt * 16 + col) * 64, lane, 0);
            acc[nt] = wmma_f16(a0, bm, acc[nt]);
        }
#pragma unroll
        for (int nt = 0; nt < 4; ++nt) {
            v16h bm = lds_load_ab(wL[o] + (nt * 16 + col) * 64, lane, 1);
            acc[nt] = wmma_f16(a1, bm, acc[nt]);
        }
#pragma unroll
        for (int nt = 0; nt < 4; ++nt) {
            int c = nt * 16 + col;
            float s = 0.f, s2 = 0.f;
#pragma unroll
            for (int t = 0; t < 8; ++t) {
                int r = r0 + rbase + t;
                outp[o][r * 64 + c] = acc[nt][t];
                s += acc[nt][t]; s2 += acc[nt][t] * acc[nt][t];
            }
            if (o < 2) {
                s  += __shfl_xor(s, 16, 32);
                s2 += __shfl_xor(s2, 16, 32);
                if (lane < 16) {
                    atomicAdd(sum1p[o] + b * 64 + c, s);
                    atomicAdd(sum2p[o] + b * 64 + c, s2);
                }
            }
        }
    }
}

__global__ void k_posstats(const float* pos, float* ws) {
    int b = blockIdx.x;
    float sp[3] = {0, 0, 0}, spp[9] = {0};
    for (int n = threadIdx.x; n < NPTS; n += blockDim.x) {
        float p[3];
        p[0] = pos[(b * NPTS + n) * 3 + 0];
        p[1] = pos[(b * NPTS + n) * 3 + 1];
        p[2] = pos[(b * NPTS + n) * 3 + 2];
#pragma unroll
        for (int d = 0; d < 3; ++d) {
            sp[d] += p[d];
#pragma unroll
            for (int e = 0; e < 3; ++e) spp[d * 3 + e] += p[d] * p[e];
        }
    }
    for (int t = 0; t < 3; ++t) atomicAdd(ws + OFF_SP + b * 3 + t, sp[t]);
    for (int t = 0; t < 9; ++t) atomicAdd(ws + OFF_SPP + b * 9 + t, spp[t]);
}

// BN1 fold: mean(h_pre)=bp1 (cancels); var = Wp1 Cov(rel) Wp1^T
__global__ void k_fin1(const float* Wp1, const float* g_p, const float* b_p, float* ws) {
    __shared__ float cov[9];
    int t = threadIdx.x;
    if (t < 9) {
        int d = t / 3, e = t % 3;
        float acc = 0.f;
        for (int b = 0; b < NB; ++b)
            acc += 2.f * (float)NPTS * ws[OFF_SPP + b * 9 + t]
                 - 2.f * ws[OFF_SP + b * 3 + d] * ws[OFF_SP + b * 3 + e];
        cov[t] = acc / ((float)NB * NPTS * NPTS);
    }
    __syncthreads();
    if (t < 3) {
        float var = 0.f;
        for (int d = 0; d < 3; ++d)
            for (int e = 0; e < 3; ++e)
                var += Wp1[t * 3 + d] * Wp1[t * 3 + e] * cov[d * 3 + e];
        float s1 = g_p[t] * rsqrtf(var + EPSV);
        for (int d = 0; d < 3; ++d) ws[OFF_A1 + t * 3 + d] = Wp1[t * 3 + d] * s1;
        ws[OFF_C1 + t] = b_p[t];
    }
}

// Sweep all (i,j): mode 0 -> sum d, d^2, q*sum_j d;  mode 1 -> k*sum_i d
__global__ __launch_bounds__(256) void k_pass1(const float* pos, const float* Wp2,
                                               const float* bp2, float* ws, int mode) {
    int gw = (blockIdx.x * blockDim.x + threadIdx.x) >> 5;  // 0..1023
    int lane = threadIdx.x & 31;
    int b = gw >> 9, r = gw & 511;
    float A1[9], c1v[3];
#pragma unroll
    for (int t = 0; t < 9; ++t) A1[t] = ws[OFF_A1 + t];
#pragma unroll
    for (int t = 0; t < 3; ++t) c1v[t] = ws[OFF_C1 + t];
    float prx = pos[(b * NPTS + r) * 3 + 0];
    float pry = pos[(b * NPTS + r) * 3 + 1];
    float prz = pos[(b * NPTS + r) * 3 + 2];
    float w2[2][3], bb[2];
#pragma unroll
    for (int u = 0; u < 2; ++u) {
        int c = lane + u * 32;
        w2[u][0] = Wp2[c * 3 + 0]; w2[u][1] = Wp2[c * 3 + 1]; w2[u][2] = Wp2[c * 3 + 2];
        bb[u] = bp2[c];
    }
    float s1a[2] = {0, 0}, s2a[2] = {0, 0}, dsm[2] = {0, 0};
    float sgn = mode ? -1.f : 1.f;
    for (int s = 0; s < NPTS; ++s) {
        float dx = sgn * (prx - pos[(b * NPTS + s) * 3 + 0]);
        float dy = sgn * (pry - pos[(b * NPTS + s) * 3 + 1]);
        float dz = sgn * (prz - pos[(b * NPTS + s) * 3 + 2]);
        float h0 = fmaxf(dx * A1[0] + dy * A1[1] + dz * A1[2] + c1v[0], 0.f);
        float h1 = fmaxf(dx * A1[3] + dy * A1[4] + dz * A1[5] + c1v[1], 0.f);
        float h2 = fmaxf(dx * A1[6] + dy * A1[7] + dz * A1[8] + c1v[2], 0.f);
#pragma unroll
        for (int u = 0; u < 2; ++u) {
            float d = h0 * w2[u][0] + h1 * w2[u][1] + h2 * w2[u][2] + bb[u];
            if (mode == 0) { s1a[u] += d; s2a[u] += d * d; }
            dsm[u] += d;
        }
    }
#pragma unroll
    for (int u = 0; u < 2; ++u) {
        int c = lane + u * 32;
        if (mode == 0) {
            atomicAdd(ws + OFF_S1 + c, s1a[u]);
            atomicAdd(ws + OFF_S2 + c, s2a[u]);
            atomicAdd(ws + OFF_SQD + c, ws[OFF_Q + (b * NPTS + r) * 64 + c] * dsm[u]);
        } else {
            atomicAdd(ws + OFF_SKD + c, ws[OFF_K + (b * NPTS + r) * 64 + c] * dsm[u]);
        }
    }
}

__global__ void k_fin2(const float* Wp2, const float* bp2,
                       const float* g_w, const float* b_w, float* ws) {
    int c = threadIdx.x;
    float M = (float)NB * NPTS * NPTS;
    float qsum = 0.f, ksum = 0.f, e2 = 0.f;
    for (int b = 0; b < NB; ++b) {
        float aq = ws[OFF_AQ + b * 64 + c], ak = ws[OFF_AK + b * 64 + c];
        qsum += aq; ksum += ak;
        e2 += (float)NPTS * ws[OFF_AQ2 + b * 64 + c]
            + (float)NPTS * ws[OFF_AK2 + b * 64 + c] - 2.f * aq * ak;
    }
    float mu = ((float)NPTS * (qsum - ksum) + ws[OFF_S1 + c]) / M;
    e2 = (e2 + 2.f * (ws[OFF_SQD + c] - ws[OFF_SKD + c]) + ws[OFF_S2 + c]) / M;
    float var = e2 - mu * mu;
    float s2 = g_w[c] * rsqrtf(var + EPSV);
    float t2 = b_w[c] - mu * s2;
    ws[OFF_S2C + c] = s2; ws[OFF_T2C + c] = t2;
    ws[OFF_W2P + c * 4 + 0] = Wp2[c * 3 + 0];
    ws[OFF_W2P + c * 4 + 1] = Wp2[c * 3 + 1];
    ws[OFF_W2P + c * 4 + 2] = Wp2[c * 3 + 2];
    ws[OFF_W2P + c * 4 + 3] = bp2[c];
    ws[OFF_W2S + c * 4 + 0] = s2 * Wp2[c * 3 + 0];
    ws[OFF_W2S + c * 4 + 1] = s2 * Wp2[c * 3 + 1];
    ws[OFF_W2S + c * 4 + 2] = s2 * Wp2[c * 3 + 2];
    ws[OFF_W2S + c * 4 + 3] = s2 * bp2[c];
}

__global__ void k_qbkb(float* ws) {
    int idx = blockIdx.x * blockDim.x + threadIdx.x;  // 65536
    int c = idx & 63;
    float s2 = ws[OFF_S2C + c], t2 = ws[OFF_T2C + c];
    ws[OFF_QB + idx] = s2 * ws[OFF_Q + idx] + t2;
    ws[OFF_KB + idx] = s2 * ws[OFF_K + idx];
}

// Fused main pass: one wave per (b,i,jchunk); flash softmax over j.
__global__ __launch_bounds__(128) void k_main(const float* pos, const float* Ww1,
        const float* bw1, const float* Ww2, const float* bw2, float* ws) {
    __shared__ __align__(16) _Float16 wW1[4096], wW2[4096];
    __shared__ float4 w2pL[64], w2sL[64];
    __shared__ float bw1L[64], bw2L[64];
    __shared__ float4 hbS[4][16];
    __shared__ __align__(16) _Float16 aT[4][1024];
    int tid = threadIdx.x;
    for (int t = tid; t < 4096; t += 128) { wW1[t] = (_Float16)Ww1[t]; wW2[t] = (_Float16)Ww2[t]; }
    if (tid < 64) {
        bw1L[tid] = bw1[tid]; bw2L[tid] = bw2[tid];
        w2pL[tid] = make_float4(ws[OFF_W2P + tid * 4 + 0], ws[OFF_W2P + tid * 4 + 1],
                                ws[OFF_W2P + tid * 4 + 2], ws[OFF_W2P + tid * 4 + 3]);
        w2sL[tid] = make_float4(ws[OFF_W2S + tid * 4 + 0], ws[OFF_W2S + tid * 4 + 1],
                                ws[OFF_W2S + tid * 4 + 2], ws[OFF_W2S + tid * 4 + 3]);
    }
    __syncthreads();
    int w = tid >> 5, lane = tid & 31;
    int gid = blockIdx.x * 4 + w;          // 0..4095
    int iidx = gid >> 2;                   // b*512+i
    int chunk = gid & 3;
    int b = iidx >> 9;
    float A1[9], c1v[3];
#pragma unroll
    for (int t = 0; t < 9; ++t) A1[t] = ws[OFF_A1 + t];
#pragma unroll
    for (int t = 0; t < 3; ++t) c1v[t] = ws[OFF_C1 + t];
    float pix = pos[iidx * 3 + 0], piy = pos[iidx * 3 + 1], piz = pos[iidx * 3 + 2];
    int off = (lane < 16) ? 0 : 8;
    int row = lane & 15;
    int rbase = 8 * (lane >> 4);
    float qbr[32];
    load_runs(ws + OFF_QB + iidx * 64, off, qbr);
    const float* kbbase = ws + OFF_KB + b * NPTS * 64;
    const float* vbase  = ws + OFF_V  + b * NPTS * 64;
    const float* posb   = pos + b * NPTS * 3;
    float mrun[4], lrun[4], arun[4];
#pragma unroll
    for (int nt = 0; nt < 4; ++nt) { mrun[nt] = -3.0e38f; lrun[nt] = 0.f; arun[nt] = 0.f; }

    // warm the L2/L1 path for the first tile's streamed rows
    {
        int j0 = chunk * 128 + row;
        __builtin_prefetch(kbbase + j0 * 64, 0, 0);
        __builtin_prefetch(vbase + j0 * 64, 0, 0);
        __builtin_prefetch(vbase + j0 * 64 + 32, 0, 0);
    }

    for (int jt = chunk * 8; jt < chunk * 8 + 8; ++jt) {
        int jbase = jt * 16;
        int j = jbase + row;
        // prefetch the next tile's streamed rows while this tile computes
        if (jt + 1 < chunk * 8 + 8) {
            int jn = j + 16;
            __builtin_prefetch(kbbase + jn * 64, 0, 0);
            __builtin_prefetch(vbase + jn * 64, 0, 0);
            __builtin_prefetch(vbase + jn * 64 + 32, 0, 0);
        }
        float dx = pix - posb[j * 3 + 0];
        float dy = piy - posb[j * 3 + 1];
        float dz = piz - posb[j * 3 + 2];
        float h0 = fmaxf(dx * A1[0] + dy * A1[1] + dz * A1[2] + c1v[0], 0.f);
        float h1 = fmaxf(dx * A1[3] + dy * A1[4] + dz * A1[5] + c1v[1], 0.f);
        float h2 = fmaxf(dx * A1[6] + dy * A1[7] + dz * A1[8] + c1v[2], 0.f);
        if (lane < 16) hbS[w][row] = make_float4(h0, h1, h2, 0.f);
        float kbr[32];
        load_runs(kbbase + j * 64, off, kbr);
        v16h a[2];
#pragma unroll
        for (int ks = 0; ks < 2; ++ks)
#pragma unroll
            for (int g = 0; g < 2; ++g)
#pragma unroll
                for (int t = 0; t < 8; ++t) {
                    int c = ks * 32 + g * 16 + off + t;
                    float4 wp = w2sL[c];
                    float pre = qbr[ks * 16 + g * 8 + t] - kbr[ks * 16 + g * 8 + t]
                              + h0 * wp.x + h1 * wp.y + h2 * wp.z + wp.w;
                    a[ks][g * 8 + t] = (_Float16)fmaxf(pre, 0.f);
                }
        // GEMM1: h2 @ Ww1^T + bw1, ReLU -> aT   (ks-outer: independent WMMAs)
        {
            v8f acc[4];
#pragma unroll
            for (int nt = 0; nt < 4; ++nt) {
                float bv0 = bw1L[nt * 16 + row];
#pragma unroll
                for (int t = 0; t < 8; ++t) acc[nt][t] = bv0;
            }
#pragma unroll
            for (int nt = 0; nt < 4; ++nt) {
                v16h bm = lds_load_ab(wW1 + (nt * 16 + row) * 64, lane, 0);
                acc[nt] = wmma_f16(a[0], bm, acc[nt]);
            }
#pragma unroll
            for (int nt = 0; nt < 4; ++nt) {
                v16h bm = lds_load_ab(wW1 + (nt * 16 + row) * 64, lane, 1);
                acc[nt] = wmma_f16(a[1], bm, acc[nt]);
            }
#pragma unroll
            for (int nt = 0; nt < 4; ++nt) {
                int c = nt * 16 + row;
#pragma unroll
                for (int t = 0; t < 8; ++t)
                    aT[w][(rbase + t) * 64 + c] = (_Float16)fmaxf(acc[nt][t], 0.f);
            }
        }
        // GEMM2: act1 @ Ww2^T + bw2 -> scores, then flash-softmax update
        {
            v16h a20 = lds_load_ab(aT[w] + row * 64, lane, 0);
            v16h a21 = lds_load_ab(aT[w] + row * 64, lane, 1);
            v8f acc[4];
#pragma unroll
            for (int nt = 0; nt < 4; ++nt) {
                float bv0 = bw2L[nt * 16 + row];
#pragma unroll
                for (int t = 0; t < 8; ++t) acc[nt][t] = bv0;
            }
#pragma unroll
            for (int nt = 0; nt < 4; ++nt) {
                v16h bm = lds_load_ab(wW2 + (nt * 16 + row) * 64, lane, 0);
                acc[nt] = wmma_f16(a20, bm, acc[nt]);
            }
#pragma unroll
            for (int nt = 0; nt < 4; ++nt) {
                v16h bm = lds_load_ab(wW2 + (nt * 16 + row) * 64, lane, 1);
                acc[nt] = wmma_f16(a21, bm, acc[nt]);
            }
#pragma unroll
            for (int nt = 0; nt < 4; ++nt) {
                int c = nt * 16 + row;
                float tm = acc[nt][0];
#pragma unroll
                for (int t = 1; t < 8; ++t) tm = fmaxf(tm, acc[nt][t]);
                tm = fmaxf(tm, __shfl_xor(tm, 16, 32));
                float mn = fmaxf(mrun[nt], tm);
                float sc = __expf(mrun[nt] - mn);
                float4 wp = w2pL[c];
                float psum = 0.f, asum = 0.f;
#pragma unroll
                for (int t = 0; t < 8; ++t) {
                    int r = rbase + t;
                    float4 hh = hbS[w][r];
                    float delta = hh.x * wp.x + hh.y * wp.y + hh.z * wp.z + wp.w;
                    float vv = vbase[(jbase + r) * 64 + c];
                    float p = __expf(acc[nt][t] - mn);
                    psum += p;
                    asum += p * (vv + delta);
                }
                psum += __shfl_xor(psum, 16, 32);
                lrun[nt] = lrun[nt] * sc + psum;
                arun[nt] = arun[nt] * sc + asum;
                mrun[nt] = mn;
            }
        }
    }
#pragma unroll
    for (int nt = 0; nt < 4; ++nt) {
        float at = arun[nt] + __shfl_xor(arun[nt], 16, 32);
        if (lane < 16) {
            int c = nt * 16 + lane;
            float* pp = ws + OFF_PART + (iidx * JSPLIT + chunk) * 192;
            pp[c] = mrun[nt];
            pp[64 + c] = lrun[nt];
            pp[128 + c] = at;
        }
    }
}

__global__ void k_combine(const float* ws, float* out) {
    int idx = blockIdx.x * blockDim.x + threadIdx.x;  // 65536
    int iidx = idx >> 6, c = idx & 63;
    const float* base = ws + OFF_PART + iidx * JSPLIT * 192;
    float M = -3.0e38f;
#pragma unroll
    for (int k = 0; k < JSPLIT; ++k) M = fmaxf(M, base[k * 192 + c]);
    float num = 0.f, den = 0.f;
#pragma unroll
    for (int k = 0; k < JSPLIT; ++k) {
        float e = __expf(base[k * 192 + c] - M);
        num += e * base[k * 192 + 128 + c];
        den += e * base[k * 192 + 64 + c];
    }
    out[idx] = num / den;
}

extern "C" void kernel_launch(void* const* d_in, const int* in_sizes, int n_in,
                              void* d_out, int out_size, void* d_ws, size_t ws_size,
                              hipStream_t stream) {
    (void)in_sizes; (void)n_in; (void)out_size; (void)ws_size;
    const float* pos = (const float*)d_in[0];
    const float* cf  = (const float*)d_in[1];
    const float* Wq  = (const float*)d_in[2];
    const float* bq  = (const float*)d_in[3];
    const float* Wk  = (const float*)d_in[4];
    const float* bk  = (const float*)d_in[5];
    const float* Wv  = (const float*)d_in[6];
    const float* bv  = (const float*)d_in[7];
    const float* Wp1 = (const float*)d_in[8];
    const float* g_p = (const float*)d_in[10];
    const float* b_p = (const float*)d_in[11];
    const float* Wp2 = (const float*)d_in[12];
    const float* bp2 = (const float*)d_in[13];
    const float* g_w = (const float*)d_in[14];
    const float* b_w = (const float*)d_in[15];
    const float* Ww1 = (const float*)d_in[16];
    const float* bw1 = (const float*)d_in[17];
    const float* Ww2 = (const float*)d_in[18];
    const float* bw2 = (const float*)d_in[19];
    float* ws  = (float*)d_ws;
    float* out = (float*)d_out;

    hipLaunchKernelGGL(k_zero,     dim3(4),    dim3(256), 0, stream, ws);
    hipLaunchKernelGGL(k_qkv,      dim3(64),   dim3(32),  0, stream, cf, Wq, bq, Wk, bk, Wv, bv, ws);
    hipLaunchKernelGGL(k_posstats, dim3(2),    dim3(256), 0, stream, pos, ws);
    hipLaunchKernelGGL(k_fin1,     dim3(1),    dim3(32),  0, stream, Wp1, g_p, b_p, ws);
    hipLaunchKernelGGL(k_pass1,    dim3(128),  dim3(256), 0, stream, pos, Wp2, bp2, ws, 0);
    hipLaunchKernelGGL(k_pass1,    dim3(128),  dim3(256), 0, stream, pos, Wp2, bp2, ws, 1);
    hipLaunchKernelGGL(k_fin2,     dim3(1),    dim3(64),  0, stream, Wp2, bp2, g_w, b_w, ws);
    hipLaunchKernelGGL(k_qbkb,     dim3(256),  dim3(256), 0, stream, ws);
    hipLaunchKernelGGL(k_main,     dim3(1024), dim3(128), 0, stream, pos, Ww1, bw1, Ww2, bw2, ws);
    hipLaunchKernelGGL(k_combine,  dim3(256),  dim3(256), 0, stream, ws, out);
}
